// RSSM_87591563034884
// MI455X (gfx1250) — matrix operand order repaired and checked by reading, not compiled, not run on hardware
//
#include <hip/hip_runtime.h>
#include <math.h>

// ---------------------------------------------------------------------------
// Problem sizes (from reference): u (B,H,L) f32, lambda (P,), C (H,P), D (H,)
// B=16, H=512, L=4096, P=512.  Output y (B,H,L) f32.
//
// y[b,h,t] = sum_{m<=t} Kr[h,t-m] * u[b,h,m] + D[h]*u[b,h,t]
// Kr[h,l]  = sum_p ( C_re[h,p]*Re(lam_p^l) - C_im[h,p]*Im(lam_p^l) )
// ---------------------------------------------------------------------------

#define Bc 16
#define Hc 512
#define Lc 4096
#define Pc 512
#define KP 4128          // padded kernel row: [0..31]=0, [32+l]=Kr[l]
#define URS 4104         // LDS u row stride in f16 elems (+8 pad: 4-bank/row skew)
#define RN 4224          // reversed kernel array length (zero tail covers d0>=-96)

typedef __attribute__((ext_vector_type(16))) _Float16 v16h;
typedef __attribute__((ext_vector_type(8)))  _Float16 v8h;
typedef __attribute__((ext_vector_type(4)))  _Float16 v4h;
typedef __attribute__((ext_vector_type(8)))  float    v8f;
typedef __attribute__((ext_vector_type(8)))  unsigned int v8u;

// ---------------------------------------------------------------------------
// Kernel A0: Vandermonde table  V[p,l] = lam_p^l  (stable via exp/log form)
// ---------------------------------------------------------------------------
__global__ __launch_bounds__(256) void rssm_vand(
    const float* __restrict__ lre, const float* __restrict__ lim,
    float* __restrict__ Vre, float* __restrict__ Vim) {
  int idx = blockIdx.x * 256 + threadIdx.x;      // idx = p*L + l
  int p = idx >> 12;
  int l = idx & (Lc - 1);
  float re = lre[p], im = lim[p];
  float logr = 0.5f * logf(re * re + im * im);   // r in [0.9,0.99) -> safe
  float th = atan2f(im, re);
  float mag = expf((float)l * logr);
  float s, c;
  sincosf((float)l * th, &s, &c);
  Vre[idx] = mag * c;
  Vim[idx] = mag * s;
}

// ---------------------------------------------------------------------------
// Kernel A1: Kr = C_re@Vre - C_im@Vim, split into f16 hi/lo, front-padded.
// Grid: (H/16)*(L/256) blocks of 256 threads.  Each thread owns one l and
// 16 h-accumulators; C tile and V tile staged in LDS.
// ---------------------------------------------------------------------------
__global__ __launch_bounds__(256) void rssm_build_k(
    const float* __restrict__ Cre, const float* __restrict__ Cim,
    const float* __restrict__ Vre, const float* __restrict__ Vim,
    _Float16* __restrict__ Khi, _Float16* __restrict__ Klo) {
  extern __shared__ char smem[];
  float* sCre = (float*)smem;                 // 16*512
  float* sCim = sCre + 16 * Pc;               // 16*512
  float* sVre = sCim + 16 * Pc;               // 32*256
  float* sVim = sVre + 32 * 256;              // 32*256

  const int tid = threadIdx.x;
  const int h0 = (blockIdx.x >> 4) << 4;      // 32 h-tiles
  const int l0 = (blockIdx.x & 15) << 8;      // 16 l-chunks of 256

  for (int i = tid; i < 16 * Pc; i += 256) {
    int hh = i >> 9, p = i & (Pc - 1);
    sCre[i] = Cre[(h0 + hh) * Pc + p];
    sCim[i] = Cim[(h0 + hh) * Pc + p];
  }

  float acc[16];
#pragma unroll
  for (int hh = 0; hh < 16; ++hh) acc[hh] = 0.f;

  for (int p0 = 0; p0 < Pc; p0 += 32) {
    __syncthreads();
    for (int i = tid; i < 32 * 256; i += 256) {
      int pp = i >> 8, ll = i & 255;
      sVre[i] = Vre[(p0 + pp) * Lc + l0 + ll];
      sVim[i] = Vim[(p0 + pp) * Lc + l0 + ll];
    }
    __syncthreads();
#pragma unroll 4
    for (int pp = 0; pp < 32; ++pp) {
      float vr = sVre[pp * 256 + tid];
      float vi = sVim[pp * 256 + tid];
#pragma unroll
      for (int hh = 0; hh < 16; ++hh) {
        acc[hh] = fmaf(sCre[hh * Pc + p0 + pp], vr, acc[hh]);
        acc[hh] = fmaf(-sCim[hh * Pc + p0 + pp], vi, acc[hh]);
      }
    }
  }

#pragma unroll
  for (int hh = 0; hh < 16; ++hh) {
    float kv = acc[hh];
    _Float16 gh = (_Float16)kv;
    _Float16 gl = (_Float16)(kv - (float)gh);
    Khi[(h0 + hh) * KP + 32 + l0 + tid] = gh;
    Klo[(h0 + hh) * KP + 32 + l0 + tid] = gl;
  }
  if (l0 == 0 && tid < 32) {                  // causal zero padding
#pragma unroll
    for (int hh = 0; hh < 16; ++hh) {
      Khi[(h0 + hh) * KP + tid] = (_Float16)0.f;
      Klo[(h0 + hh) * KP + tid] = (_Float16)0.f;
    }
  }
}

// ---------------------------------------------------------------------------
// Kernel C: causal convolution as Toeplitz WMMA, QUAD-tile + ring-slot version.
// One block per h (512 blocks, 256 thr = 8 waves). Whole u[:,h,:] slice split
// to f16 hi/lo in LDS once; kernel row staged reversed in two phase copies so
// Toeplitz B-operand dword gathers are 4B aligned.
//
// Each wave owns FOUR 16x16 output tiles t0+{0,32,64,96}; m0 walks down by 32.
//   - A (u operand) depends only on m0 -> shared by all 4 tiles.
//   - B (Toeplitz) depends only on d0=t0-m0 -> each fresh B is consumed 4x as
//     it ages through a ring of 4 statically-named slots (manual 4x unroll ->
//     rotation is pure register renaming: no movs, no WMMA WAR-hazard nops).
//   - Iteration count padded to a multiple of 4 by starting m0 above m0max:
//     those steps read the extended ZERO tail of the reversed kernel arrays
//     (d0<0 -> B==0), so extra/inactive contributions vanish exactly with no
//     branches (EXEC stays all-ones, as WMMA requires).
// Steady state: 4x ds_b128 + 8 dual-dword ds loads per 12 WMMAs (1 DS/wmma).
// Precision: c += A_hi x B_hi + A_hi x B_lo + A_lo x B_hi  (~fp32 accurate)
// ---------------------------------------------------------------------------

// One pipeline step: load A(m0) and fresh B (tile d), fire 12 WMMAs.
// Oldest slots first so the fresh ds loads have ~4 WMMAs of latency cover.
#define CONV_STEP(M0, BNH, BNL, B1H, B1L, B2H, B2L, B3H, B3L)                  \
  do {                                                                         \
    const int m0_ = (M0);                                                      \
    const v8h a0h = *(const v8h*)(uhs + arow + m0_ + kb);                      \
    const v8h a1h = *(const v8h*)(uhs + arow + m0_ + 8 + kb);                  \
    const v8h a0l = *(const v8h*)(uls + arow + m0_ + kb);                      \
    const v8h a1l = *(const v8h*)(uls + arow + m0_ + 8 + kb);                  \
    v16h Ah = __builtin_shufflevector(a0h, a1h, 0,1,2,3,4,5,6,7,8,9,10,11,12,13,14,15); \
    v16h Al = __builtin_shufflevector(a0l, a1l, 0,1,2,3,4,5,6,7,8,9,10,11,12,13,14,15); \
    const int ebase = (sfix - (t0d - m0_)) & ~1;                               \
    v8u wh_, wl_;                                                              \
    _Pragma("unroll")                                                          \
    for (int v = 0; v < 8; ++v) {                                              \
      wh_[v] = *(const unsigned int*)(bh_base + ebase + 2 * v);                \
      wl_[v] = *(const unsigned int*)(bl_base + ebase + 2 * v);                \
    }                                                                          \
    BNH = __builtin_bit_cast(v16h, wh_);                                       \
    BNL = __builtin_bit_cast(v16h, wl_);                                       \
    c0 = __builtin_amdgcn_wmma_f32_16x16x32_f16(false, Ah, false, B3H, (short)0, c0, false, false); \
    c1 = __builtin_amdgcn_wmma_f32_16x16x32_f16(false, Ah, false, B2H, (short)0, c1, false, false); \
    c2 = __builtin_amdgcn_wmma_f32_16x16x32_f16(false, Ah, false, B1H, (short)0, c2, false, false); \
    c3 = __builtin_amdgcn_wmma_f32_16x16x32_f16(false, Ah, false, BNH, (short)0, c3, false, false); \
    c0 = __builtin_amdgcn_wmma_f32_16x16x32_f16(false, Ah, false, B3L, (short)0, c0, false, false); \
    c1 = __builtin_amdgcn_wmma_f32_16x16x32_f16(false, Ah, false, B2L, (short)0, c1, false, false); \
    c2 = __builtin_amdgcn_wmma_f32_16x16x32_f16(false, Ah, false, B1L, (short)0, c2, false, false); \
    c3 = __builtin_amdgcn_wmma_f32_16x16x32_f16(false, Ah, false, BNL, (short)0, c3, false, false); \
    c0 = __builtin_amdgcn_wmma_f32_16x16x32_f16(false, Al, false, B3H, (short)0, c0, false, false); \
    c1 = __builtin_amdgcn_wmma_f32_16x16x32_f16(false, Al, false, B2H, (short)0, c1, false, false); \
    c2 = __builtin_amdgcn_wmma_f32_16x16x32_f16(false, Al, false, B1H, (short)0, c2, false, false); \
    c3 = __builtin_amdgcn_wmma_f32_16x16x32_f16(false, Al, false, BNH, (short)0, c3, false, false); \
  } while (0)

__global__ __launch_bounds__(256) void rssm_conv(
    const float* __restrict__ u,
    const _Float16* __restrict__ Khi, const _Float16* __restrict__ Klo,
    const float* __restrict__ Dsk, float* __restrict__ y) {
  extern __shared__ char smem[];
  _Float16* uhs = (_Float16*)smem;            // [16][URS]
  _Float16* uls = uhs + Bc * URS;             // [16][URS]
  _Float16* Rh  = uls + Bc * URS;             // even-phase reversed K_hi
  _Float16* Oh  = Rh + RN;                    // odd-phase  (Oh[x] = Rh[x+1])
  _Float16* Rl  = Oh + RN;
  _Float16* Ol  = Rl + RN;

  const int h = blockIdx.x;
  const int tid = threadIdx.x;

  // Stage reversed kernel rows: R[x] = Kpad[4127 - x], O[x] = R[x+1]
  // (x > 4127 -> zero tail; also provides B==0 for the padded d0<0 steps)
  const _Float16* krh = Khi + (size_t)h * KP;
  const _Float16* krl = Klo + (size_t)h * KP;
  for (int x = tid; x < RN; x += 256) {
    int s0 = 4127 - x;
    int s1 = 4126 - x;
    Rh[x] = (s0 >= 0) ? krh[s0] : (_Float16)0.f;
    Oh[x] = (s1 >= 0) ? krh[s1] : (_Float16)0.f;
    Rl[x] = (s0 >= 0) ? krl[s0] : (_Float16)0.f;
    Ol[x] = (s1 >= 0) ? krl[s1] : (_Float16)0.f;
  }
  // Stage u[:,h,:] split hi/lo, 4 elems/thread/iter (b128 loads, b64 stores)
  for (int i = tid * 4; i < Bc * Lc; i += 256 * 4) {
    int b = i >> 12, m = i & (Lc - 1);
    const float4 g = *(const float4*)(u + ((size_t)(b * Hc + h)) * Lc + m);
    float gv[4] = {g.x, g.y, g.z, g.w};
    v4h hp, lp;
#pragma unroll
    for (int k = 0; k < 4; ++k) {
      _Float16 gh = (_Float16)gv[k];
      hp[k] = gh;
      lp[k] = (_Float16)(gv[k] - (float)gh);
    }
    *(v4h*)(uhs + b * URS + m) = hp;
    *(v4h*)(uls + b * URS + m) = lp;
  }
  __syncthreads();

  const float dh = Dsk[h];
  const int lane = tid & 31;
  const int l15 = lane & 15;                  // A row (=batch) and B col (=n)
  const int hl  = lane >> 4;
  const int kb  = hl * 16;                    // A-layout K sub-block
  const int wave = tid >> 5;
  const int arow = l15 * URS;
  // Per-lane parity of the Toeplitz gather index is constant -> pick phase once
  const _Float16* bh_base = (l15 & 1) ? Rh : Oh;
  const _Float16* bl_base = (l15 & 1) ? Rl : Ol;
  const int sfix = 4095 - l15 + 16 * hl;      // sbase = sfix - d0

  // 64 tile-quads: group g of 8 tiles {8g..8g+7} -> quads
  // (8g,8g+2,8g+4,8g+6) and (8g+1,8g+3,8g+5,8g+7); round-robin over waves.
  for (int qq = wave; qq < 64; qq += 8) {
    const int tta = ((qq >> 1) << 3) + (qq & 1);
    const int t0a = tta << 4;                 // tiles at t0a + {0,32,64,96}
    const int t0d = t0a + 96;
    const int m0max_d = ((t0d + 15) >> 5) << 5;
    // Pad trip count to a multiple of 4 (extra high-m0 steps hit zero B)
    const int tc = (m0max_d >> 5) + 1;
    const int m0start = m0max_d + 32 * ((4 - (tc & 3)) & 3);  // == 96 mod 128

    v8f c0 = {}, c1 = {}, c2 = {}, c3 = {};   // tiles a,b,c,d
    v16h S0h = {}, S0l = {}, S1h = {}, S1l = {};
    v16h S2h = {}, S2l = {}, S3h = {}, S3l = {};

#pragma clang loop unroll(disable)
    for (int m0 = m0start; m0 >= 96; m0 -= 128) {
      CONV_STEP(m0,      S0h, S0l, S3h, S3l, S2h, S2l, S1h, S1l);
      CONV_STEP(m0 - 32, S1h, S1l, S0h, S0l, S3h, S3l, S2h, S2l);
      CONV_STEP(m0 - 64, S2h, S2l, S1h, S1l, S0h, S0l, S3h, S3l);
      CONV_STEP(m0 - 96, S3h, S3l, S2h, S2l, S1h, S1l, S0h, S0l);
    }

    // Epilogue: D tile lanes 0-15 col n=lane (time), rows M=vgpr (+8 lanes>=16)
    const v8f cs[4] = {c0, c1, c2, c3};
#pragma unroll
    for (int q = 0; q < 4; ++q) {
      const int t = t0a + 32 * q + l15;
#pragma unroll
      for (int v = 0; v < 8; ++v) {
        int b = v + 8 * hl;
        float uv = (float)uhs[b * URS + t] + (float)uls[b * URS + t];
        y[((size_t)(b * Hc + h)) * Lc + t] = cs[q][v] + dh * uv;
      }
    }
  }
}

// ---------------------------------------------------------------------------
// Launch: A0 (Vandermonde) -> A1 (kernel build+split) -> C (WMMA conv)
// Workspace: Vre (8MB) | Vim (8MB) | Khi (4.13MB) | Klo (4.13MB) ~ 24.5MB
// ---------------------------------------------------------------------------
extern "C" void kernel_launch(void* const* d_in, const int* in_sizes, int n_in,
                              void* d_out, int out_size, void* d_ws, size_t ws_size,
                              hipStream_t stream) {
  (void)in_sizes; (void)n_in; (void)out_size; (void)ws_size;
  const float* u   = (const float*)d_in[0];
  const float* lre = (const float*)d_in[1];
  const float* lim = (const float*)d_in[2];
  const float* Cre = (const float*)d_in[3];
  const float* Cim = (const float*)d_in[4];
  const float* Dsk = (const float*)d_in[5];
  float* y = (float*)d_out;

  char* ws = (char*)d_ws;
  float*     Vre = (float*)ws;
  float*     Vim = Vre + (size_t)Pc * Lc;
  _Float16*  Khi = (_Float16*)(Vim + (size_t)Pc * Lc);
  _Float16*  Klo = Khi + (size_t)Hc * KP;

  // A0: 512*4096/256 = 8192 blocks
  rssm_vand<<<dim3((Pc * Lc) / 256), dim3(256), 0, stream>>>(lre, lim, Vre, Vim);

  // A1: (512/16)*(4096/256) = 512 blocks; 128KB dynamic LDS
  const int ldsA1 = (16 * Pc * 2 + 32 * 256 * 2) * (int)sizeof(float);
  rssm_build_k<<<dim3((Hc / 16) * (Lc / 256)), dim3(256), ldsA1, stream>>>(
      Cre, Cim, Vre, Vim, Khi, Klo);

  // C: 512 blocks (one per h); ~290KB dynamic LDS (fits 320KB/WGP)
  const int ldsC = (2 * Bc * URS + 4 * RN) * (int)sizeof(_Float16);
  rssm_conv<<<dim3(Hc), dim3(256), ldsC, stream>>>(u, Khi, Klo, Dsk, y);
}
